// MultiHeadAttention_80393197846928
// MI455X (gfx1250) — compile-verified, matrix operand order checked by reading, not run on hardware
//
#include <hip/hip_runtime.h>
#include <hip/hip_bf16.h>

typedef __attribute__((ext_vector_type(16))) _Float16 v16h;
typedef __attribute__((ext_vector_type(8)))  _Float16 h8;
typedef __attribute__((ext_vector_type(8)))  float    v8f;
typedef int i32x4 __attribute__((vector_size(16)));

union Frag16 { v16h v; h8 h[2]; };

constexpr int E  = 1024;
constexpr int S  = 2048;
constexpr int H  = 16;
constexpr int DH = 64;
constexpr int NB = 2;

#define AS1 __attribute__((address_space(1)))
#define AS3 __attribute__((address_space(3)))

#if __has_builtin(__builtin_amdgcn_global_load_async_to_lds_b128)
#define FA_ASYNC 1
#define FA_BUFS  2
#else
#define FA_ASYNC 0
#define FA_BUFS  1
#endif

__device__ inline void async_wait0() {
#if __has_builtin(__builtin_amdgcn_s_wait_asynccnt)
    __builtin_amdgcn_s_wait_asynccnt(0);
#elif FA_ASYNC
    asm volatile("s_wait_asynccnt 0x0" ::: "memory");
#endif
}

#if FA_ASYNC
// 16B global -> LDS async copy (GLOBAL_LOAD_ASYNC_TO_LDS_B128, ASYNCcnt).
// Builtin signature (from hipcc diagnostic): (AS1 i32x4*, AS3 i32x4*, int, int)
__device__ inline void cp_async_b128(const _Float16* g, _Float16* l) {
    __builtin_amdgcn_global_load_async_to_lds_b128(
        (AS1 i32x4*)(AS1 void*)const_cast<_Float16*>(g),
        (AS3 i32x4*)(AS3 void*)l, 0, 0);
}
#endif

// Load a 16x32 f16 WMMA operand fragment: row pointer (32 contiguous f16 of the
// K-dim), lane-group g selects the documented K distribution:
//   g=0: V0-3 = K0..7,  V4-7 = K16..23
//   g=1: V0-3 = K8..15, V4-7 = K24..31
__device__ inline v16h load_frag32(const _Float16* rowp, int g) {
    Frag16 u;
    u.h[0] = *(const h8*)(rowp + g * 8);
    u.h[1] = *(const h8*)(rowp + 16 + g * 8);
    return u.v;
}

__device__ inline v8f wmma_f16(v16h a, v16h b, v8f c) {
    return __builtin_amdgcn_wmma_f32_16x16x32_f16(false, a, false, b,
                                                  (short)0, c, false, false);
}

// ---------------------------------------------------------------------------
// Projection GEMM: Out(o,s) = W(o,e) * X(e,s) + bias(o), scaled.
// MODE 0: f16 out, per-head q-major  [b][h][s][64]   (Q with scale, K)
// MODE 1: f16 out, per-head c-major  [b][h][64][s]   (V)
// MODE 2: f32 out, (B,E,1,S) layout                  (final O projection)
// XT = float (X is (B,E,1,S) f32) or _Float16 (X is attnT [B][S][E] f16)
// ---------------------------------------------------------------------------
template <int MODE, typename XT>
__global__ __launch_bounds__(256)
void proj_gemm(const float* __restrict__ W, const float* __restrict__ bias,
               const XT* __restrict__ X, void* __restrict__ outp, float scale) {
    __shared__ _Float16 At[128][40];   // W tile:  o-local x k   (row stride 80B)
    __shared__ _Float16 Bt[128][40];   // X tile:  s-local x k

    const int t    = threadIdx.x;
    const int lane = t & 31;
    const int wv   = t >> 5;
    const int g    = lane >> 4;
    const int ln   = lane & 15;
    const int ob   = blockIdx.x * 128;
    const int sb   = blockIdx.y * 128;
    const int bz   = blockIdx.z;
    const int wm   = wv & 1;    // 2 m-subtiles of 64
    const int wn   = wv >> 1;   // 4 n-subtiles of 32

    const v8f vzero = {0.f, 0.f, 0.f, 0.f, 0.f, 0.f, 0.f, 0.f};
    v8f acc[4][2];
#pragma unroll
    for (int mi = 0; mi < 4; ++mi)
#pragma unroll
        for (int ni = 0; ni < 2; ++ni) acc[mi][ni] = vzero;

    for (int kb = 0; kb < E; kb += 32) {
        __syncthreads();
        {   // stage A (W): each thread converts 16 f32 -> f16, 2x b128 LDS stores
            const int r  = t >> 1;
            const int c0 = (t & 1) * 16;
            const float4* src = (const float4*)(W + (size_t)(ob + r) * E + kb + c0);
#pragma unroll
            for (int j = 0; j < 2; ++j) {
                float4 f0 = src[2 * j];
                float4 f1 = src[2 * j + 1];
                h8 hv;
                hv[0] = (_Float16)f0.x; hv[1] = (_Float16)f0.y;
                hv[2] = (_Float16)f0.z; hv[3] = (_Float16)f0.w;
                hv[4] = (_Float16)f1.x; hv[5] = (_Float16)f1.y;
                hv[6] = (_Float16)f1.z; hv[7] = (_Float16)f1.w;
                *(h8*)&At[r][c0 + j * 8] = hv;
            }
        }
        if constexpr (sizeof(XT) == 4) {
            // X is f32 (B,E,1,S): transpose into Bt[s][k] during staging
            const int k  = t >> 3;
            const int s0 = (t & 7) * 16;
            const float4* src = (const float4*)((const float*)X +
                    (size_t)bz * E * S + (size_t)(kb + k) * S + sb + s0);
#pragma unroll
            for (int j = 0; j < 4; ++j) {
                float4 f = src[j];
                Bt[s0 + 4 * j + 0][k] = (_Float16)f.x;
                Bt[s0 + 4 * j + 1][k] = (_Float16)f.y;
                Bt[s0 + 4 * j + 2][k] = (_Float16)f.z;
                Bt[s0 + 4 * j + 3][k] = (_Float16)f.w;
            }
        } else {
            // X is f16 attnT [B][S][E]: s-major and k-contiguous -> direct
            // global->LDS async copies (no VGPR bounce) when available.
            const int s  = t >> 1;
            const int c0 = (t & 1) * 16;
            const _Float16* src = (const _Float16*)X +
                    ((size_t)bz * S + sb + s) * E + kb + c0;
#if FA_ASYNC
            cp_async_b128(src,     &Bt[s][c0]);
            cp_async_b128(src + 8, &Bt[s][c0 + 8]);
            async_wait0();
#else
            *(h8*)&Bt[s][c0]     = ((const h8*)src)[0];
            *(h8*)&Bt[s][c0 + 8] = ((const h8*)src)[1];
#endif
        }
        __syncthreads();

        v16h af[4], bf[2];
#pragma unroll
        for (int mi = 0; mi < 4; ++mi)
            af[mi] = load_frag32(&At[wm * 64 + mi * 16 + ln][0], g);
#pragma unroll
        for (int ni = 0; ni < 2; ++ni)
            bf[ni] = load_frag32(&Bt[wn * 32 + ni * 16 + ln][0], g);
#pragma unroll
        for (int mi = 0; mi < 4; ++mi)
#pragma unroll
            for (int ni = 0; ni < 2; ++ni)
                acc[mi][ni] = wmma_f16(af[mi], bf[ni], acc[mi][ni]);
    }

    // Epilogue: D layout -> row m = g*8 + r, col n = ln
#pragma unroll
    for (int mi = 0; mi < 4; ++mi) {
#pragma unroll
        for (int ni = 0; ni < 2; ++ni) {
#pragma unroll
            for (int r = 0; r < 8; ++r) {
                const int o = ob + wm * 64 + mi * 16 + g * 8 + r;
                const int s = sb + wn * 32 + ni * 16 + ln;
                const float val = (acc[mi][ni][r] + bias[o]) * scale;
                if constexpr (MODE == 0) {
                    _Float16* out = (_Float16*)outp;
                    out[(((size_t)bz * H + (o >> 6)) * S + s) * DH + (o & 63)] =
                        (_Float16)val;
                } else if constexpr (MODE == 1) {
                    _Float16* out = (_Float16*)outp;
                    out[(((size_t)bz * H + (o >> 6)) * DH + (o & 63)) * S + s] =
                        (_Float16)val;
                } else {
                    float* out = (float*)outp;
                    out[((size_t)bz * E + o) * S + s] = val;
                }
            }
        }
    }
}

// ---------------------------------------------------------------------------
// Flash attention: per (b, h, 128-q block). 8 waves, each wave owns 16 q cols.
// K/V tiles (32 k-steps) double-buffered in LDS via GLOBAL_LOAD_ASYNC_TO_LDS:
// the async engine fills buffer i+1 while WMMAs consume buffer i.
// scores(k,q) = sum_c Kh[k][c] * Qh[q][c] ; softmax over k ; out = P * V.
// ---------------------------------------------------------------------------
__global__ __launch_bounds__(256)
void flash_attn(const _Float16* __restrict__ Qh, const _Float16* __restrict__ Kh,
                const _Float16* __restrict__ Vh, const float* __restrict__ mask,
                _Float16* __restrict__ attnT) {
    __shared__ _Float16 Kt[FA_BUFS][32][72];   // k-local x c(64), stride 144B
    __shared__ _Float16 Vt[FA_BUFS][64][40];   // c x k-local(32), stride 80B

    const int t    = threadIdx.x;
    const int lane = t & 31;
    const int wv   = t >> 5;
    const int g    = lane >> 4;
    const int ln   = lane & 15;
    const int qblk = blockIdx.x;
    const int h    = blockIdx.y;
    const int bz   = blockIdx.z;
    const size_t bh = (size_t)bz * H + h;
    const int qcol = qblk * 128 + wv * 16 + ln;

    const _Float16* kbase = Kh + bh * (size_t)S * DH;
    const _Float16* vbase = Vh + bh * (size_t)DH * S;
    const float*    mbase = mask + (size_t)bz * S * S;

    // Q fragments for the two 32-wide c blocks, kept in registers
    const _Float16* qp = Qh + (bh * S + qcol) * DH;
    v16h bq[2];
#pragma unroll
    for (int cb = 0; cb < 2; ++cb) {
        Frag16 u;
        u.h[0] = *(const h8*)(qp + cb * 32 + g * 8);
        u.h[1] = *(const h8*)(qp + cb * 32 + 16 + g * 8);
        bq[cb] = u.v;
    }

    const v8f vzero = {0.f, 0.f, 0.f, 0.f, 0.f, 0.f, 0.f, 0.f};
    v8f acc[4];
#pragma unroll
    for (int ct = 0; ct < 4; ++ct) acc[ct] = vzero;
    float m_run = -3.0e38f;
    float l_run = 0.f;

    // Per-thread staging slices: one b128 of K tile, one b128 of V tile.
    const int kr = t >> 3, kc0 = (t & 7) * 8;
    const int vc = t >> 2, vk0 = (t & 3) * 8;
    auto stage = [&](int buf, int kb) {
#if FA_ASYNC
        cp_async_b128(kbase + (size_t)(kb + kr) * DH + kc0, &Kt[buf][kr][kc0]);
        cp_async_b128(vbase + (size_t)vc * S + kb + vk0,    &Vt[buf][vc][vk0]);
#else
        *(h8*)&Kt[buf][kr][kc0] = *(const h8*)(kbase + (size_t)(kb + kr) * DH + kc0);
        *(h8*)&Vt[buf][vc][vk0] = *(const h8*)(vbase + (size_t)vc * S + kb + vk0);
#endif
    };

    constexpr int NIT = S / 32;
#if FA_ASYNC
    stage(0, 0);   // prologue: tile 0 in flight before the loop
#endif
    for (int i = 0; i < NIT; ++i) {
        const int kb  = i * 32;
        const int buf = i & (FA_BUFS - 1);
#if FA_ASYNC
        async_wait0();        // this wave's copies (tile i) landed in LDS
        __syncthreads();      // all waves' copies landed; buf^1 free for reuse
        if (i + 1 < NIT) stage(buf ^ 1, kb + 32);   // overlap with compute
#else
        __syncthreads();
        stage(0, kb);
        __syncthreads();
#endif
        // Prefetch next iteration's mask rows (global_prefetch_b8)
        if (kb + 32 < S) {
            __builtin_prefetch(mbase + (size_t)(kb + 32 + g * 8) * S + qcol, 0, 0);
            __builtin_prefetch(mbase + (size_t)(kb + 48 + g * 8) * S + qcol, 0, 0);
        }

        // --- scores: two 16x16 D tiles covering k-local 0..15 and 16..31 ---
        v8f s0 = vzero, s1 = vzero;
#pragma unroll
        for (int cb = 0; cb < 2; ++cb) {
            v16h a0 = load_frag32(&Kt[buf][ln][cb * 32], g);
            v16h a1 = load_frag32(&Kt[buf][16 + ln][cb * 32], g);
            s0 = wmma_f16(a0, bq[cb], s0);
            s1 = wmma_f16(a1, bq[cb], s1);
        }
        // mask add: lane holds column q = qcol, rows k = kb + {g*8+r, 16+g*8+r}
#pragma unroll
        for (int r = 0; r < 8; ++r) {
            s0[r] += mbase[(size_t)(kb + g * 8 + r) * S + qcol];
            s1[r] += mbase[(size_t)(kb + 16 + g * 8 + r) * S + qcol];
        }
        // --- online softmax over k (per q = per lane + one xor-16 shuffle) ---
        float mx = s0[0];
#pragma unroll
        for (int r = 1; r < 8; ++r) mx = fmaxf(mx, s0[r]);
#pragma unroll
        for (int r = 0; r < 8; ++r) mx = fmaxf(mx, s1[r]);
        mx = fmaxf(mx, __shfl_xor(mx, 16));
        const float mnew = fmaxf(m_run, mx);
        const float corr = __expf(m_run - mnew);

        float p0[8], p1[8], rs = 0.f;
#pragma unroll
        for (int r = 0; r < 8; ++r) {
            p0[r] = __expf(s0[r] - mnew);
            p1[r] = __expf(s1[r] - mnew);
            rs += p0[r] + p1[r];
        }
        rs += __shfl_xor(rs, 16);
        l_run = l_run * corr + rs;
        m_run = mnew;
#pragma unroll
        for (int ct = 0; ct < 4; ++ct) acc[ct] *= corr;

        // D layout feeds the PV B-operand directly: h[0] = k 0..15 half,
        // h[1] = k 16..31 half (per lane-group K distribution).
        Frag16 w;
#pragma unroll
        for (int j = 0; j < 8; ++j) {
            w.h[0][j] = (_Float16)p0[j];
            w.h[1][j] = (_Float16)p1[j];
        }

        // --- PV: acc(c,q) += V(c,k) * P(k,q), 4 c-subtiles of 16 ---
#pragma unroll
        for (int ct = 0; ct < 4; ++ct) {
            v16h av = load_frag32(&Vt[buf][ct * 16 + ln][0], g);
            acc[ct] = wmma_f16(av, w.v, acc[ct]);
        }
    }

    const float inv = 1.0f / l_run;
#pragma unroll
    for (int ct = 0; ct < 4; ++ct) {
#pragma unroll
        for (int r = 0; r < 8; ++r) {
            const int cg = h * DH + ct * 16 + g * 8 + r;
            attnT[((size_t)bz * S + qcol) * E + cg] = (_Float16)(acc[ct][r] * inv);
        }
    }
}

// ---------------------------------------------------------------------------
extern "C" void kernel_launch(void* const* d_in, const int* in_sizes, int n_in,
                              void* d_out, int out_size, void* d_ws, size_t ws_size,
                              hipStream_t stream) {
    const float* q    = (const float*)d_in[0];
    const float* k    = (const float*)d_in[1];
    const float* v    = (const float*)d_in[2];
    const float* mask = (const float*)d_in[3];
    const float* Wq   = (const float*)d_in[4];
    const float* bq   = (const float*)d_in[5];
    const float* Wk   = (const float*)d_in[6];
    const float* bk   = (const float*)d_in[7];
    const float* Wv   = (const float*)d_in[8];
    const float* bv   = (const float*)d_in[9];
    const float* Wo   = (const float*)d_in[10];
    const float* bo   = (const float*)d_in[11];

    const size_t act_bytes = (size_t)NB * S * E * sizeof(_Float16);  // 8 MB
    char* ws = (char*)d_ws;
    _Float16* Qh16  = (_Float16*)(ws + 0 * act_bytes);  // [b][h][s][64]
    _Float16* Kh16  = (_Float16*)(ws + 1 * act_bytes);  // [b][h][s][64]
    _Float16* Vh16  = (_Float16*)(ws + 2 * act_bytes);  // [b][h][64][s]
    _Float16* attnT = (_Float16*)(ws + 3 * act_bytes);  // [b][s][E]

    const float scale = 0.125f;  // 1/sqrt(dh=64)
    const dim3 blk(256);
    const dim3 gproj(E / 128, S / 128, NB);
    const dim3 gattn(S / 128, H, NB);

    proj_gemm<0, float><<<gproj, blk, 0, stream>>>(Wq, bq, q, (void*)Qh16, scale);
    proj_gemm<0, float><<<gproj, blk, 0, stream>>>(Wk, bk, k, (void*)Kh16, 1.0f);
    proj_gemm<1, float><<<gproj, blk, 0, stream>>>(Wv, bv, v, (void*)Vh16, 1.0f);
    flash_attn<<<gattn, blk, 0, stream>>>(Qh16, Kh16, Vh16, mask, attnT);
    proj_gemm<2, _Float16><<<gproj, blk, 0, stream>>>(Wo, bo, attnT, d_out, 1.0f);
}
